// Head_57604101374456
// MI455X (gfx1250) — matrix operand order repaired and checked by reading, not compiled, not run on hardware
//
#include <hip/hip_runtime.h>
#include <hip/hip_fp16.h>
#include <math.h>

typedef __attribute__((ext_vector_type(16))) _Float16 v16h;
typedef __attribute__((ext_vector_type(8)))  _Float16 v8h;
typedef __attribute__((ext_vector_type(8)))  float    v8f;

#define BB 512
#define TT 256
#define CC 384
#define HH 64
#define WAVES 16
#define NTHREADS (WAVES * 32)

// f16 per lane-slot in LDS fragment buffers: 24 f16 = 48 B -> 16B-aligned for b128,
// dword-stride 12 across lanes -> bank-balanced.
#define FS 24

// reference multiplies by C**0.5 (faithful to original source)
#define ATTN_SCALE 19.595917942265423f

// W fragment image in workspace: [kk(12)][m(3)][nt(4)][lane(32)][16] f16 (32B/lane slot)
#define WF_TOTAL (12 * 3 * 4 * 32 * 16)   // 73728 f16 = 147456 bytes

// ---------------- LDS map (f16 units) ----------------
#define QF_OFF 0                              // Q A-frags   [wave(16)][f(2)][lane(32)]*FS
#define KF_OFF (QF_OFF + 16 * 2 * 32 * FS)    // K^T B-frags [kt(16)][f(2)][lane(32)]*FS
#define VF_OFF (KF_OFF + 16 * 2 * 32 * FS)    // V B-frags   [jb(8)][nt(4)][lane(32)]*FS
#define PS_OFF (VF_OFF + 8 * 4 * 32 * FS)     // P A-frags   [wave(16)][lane(32)]*FS
#define SMEM_HALVES (PS_OFF + 16 * 32 * FS)
#define SMEM_BYTES ((size_t)SMEM_HALVES * 2)  // ~168 KB of the 320 KB WGP pool

static __device__ __forceinline__ v8f v8f_zero() {
  v8f z;
#pragma unroll
  for (int i = 0; i < 8; ++i) z[i] = 0.0f;
  return z;
}

// 16-element f16 fragment from two 16B halves (2x b128, LDS or global)
static __device__ __forceinline__ v16h load_frag(const _Float16* p) {
  union { v16h v; v8h h[2]; } u;
  u.h[0] = *(const v8h*)p;
  u.h[1] = *(const v8h*)(p + 8);
  return u.v;
}

// ---------- pre-pass: build f16 W fragment image (all K-slabs) in workspace ----------
// frag element (kk, m, nt, lane=(hk,n), i) = W_m[kk*32 + (i>>3)*16 + hk*8 + (i&7)][nt*16 + n]
extern "C" __global__ __launch_bounds__(256)
void build_wfrags(const float* __restrict__ Wq, const float* __restrict__ Wk,
                  const float* __restrict__ Wv, _Float16* __restrict__ wf) {
  int idx = blockIdx.x * 256 + threadIdx.x;
  if (idx >= WF_TOTAL) return;
  int i    = idx & 15;
  int lane = (idx >> 4) & 31;
  int nt   = (idx >> 9) & 3;
  int m    = (idx >> 11) % 3;
  int kk   = idx / (3 * 4 * 32 * 16);
  int hk = lane >> 4, n = lane & 15;
  const float* wp = (m == 0) ? Wq : ((m == 1) ? Wk : Wv);
  wf[idx] = (_Float16)wp[(kk * 32 + (i >> 3) * 16 + hk * 8 + (i & 7)) * HH + nt * 16 + n];
}

extern "C" __global__ __launch_bounds__(NTHREADS)
void attn_head_fused(const float* __restrict__ x,
                     const _Float16* __restrict__ wf,
                     float* __restrict__ out) {
  extern __shared__ _Float16 smem[];
  const int b    = blockIdx.x;
  const int tid  = threadIdx.x;
  const int wave = tid >> 5;
  const int lane = tid & 31;
  const int half = lane >> 4;   // 16-lane half-group
  const int l16  = lane & 15;

  const float* xb = x + (size_t)b * TT * CC;

  // -------- Phase 1: q,k,v = x @ W  (barrier-free; A from global x, B from W-frag image) ----
  v8f acc[3][4];
#pragma unroll
  for (int m = 0; m < 3; ++m)
#pragma unroll
    for (int nt = 0; nt < 4; ++nt) acc[m][nt] = v8f_zero();

  for (int kk = 0; kk < CC / 32; ++kk) {
    // A-fragment straight from global x: row = wave*16 + l16 (private to this wave),
    // per-lane data = cols {half*8+0..7, half*8+16..23} -> four aligned float4 loads
    const float* xr = xb + (wave * 16 + l16) * CC + kk * 32 + half * 8;
    float4 x0 = *(const float4*)(xr);
    float4 x1 = *(const float4*)(xr + 4);
    float4 x2 = *(const float4*)(xr + 16);
    float4 x3 = *(const float4*)(xr + 20);
    v16h a;
    a[0] = (_Float16)x0.x; a[1] = (_Float16)x0.y; a[2]  = (_Float16)x0.z; a[3]  = (_Float16)x0.w;
    a[4] = (_Float16)x1.x; a[5] = (_Float16)x1.y; a[6]  = (_Float16)x1.z; a[7]  = (_Float16)x1.w;
    a[8] = (_Float16)x2.x; a[9] = (_Float16)x2.y; a[10] = (_Float16)x2.z; a[11] = (_Float16)x2.w;
    a[12]= (_Float16)x3.x; a[13]= (_Float16)x3.y; a[14] = (_Float16)x3.z; a[15] = (_Float16)x3.w;

    const _Float16* wkk = wf + (size_t)kk * (3 * 4 * 32 * 16);
#pragma unroll
    for (int m = 0; m < 3; ++m) {
#pragma unroll
      for (int nt = 0; nt < 4; ++nt) {
        v16h bf = load_frag(wkk + ((m * 4 + nt) * 32 + lane) * 16);  // L2-resident
        acc[m][nt] = __builtin_amdgcn_wmma_f32_16x16x32_f16(
            false, a, false, bf, (short)0, acc[m][nt], false, false);
      }
    }
  }

  // ---- projection epilogue: scatter accumulators straight into LDS fragment layouts ----
  // C/D layout: vgpr j, lane (half,l16) -> row r = j + 8*half, col = nt*16 + l16
#pragma unroll
  for (int nt = 0; nt < 4; ++nt)
#pragma unroll
    for (int j = 0; j < 8; ++j) {
      int r = j + 8 * half;
      // Q -> A-frag layout (read back by this wave only)
      smem[QF_OFF + ((wave * 2 + (nt >> 1)) * 32 + (l16 >> 3) * 16 + r) * FS +
           (l16 & 7) + (nt & 1) * 8] = (_Float16)acc[0][nt][j];
      // K -> K^T B-frag layout: frag(kt=wave, f) element (lane=(hk,n), i) = k[kt*16+n][f*32+(i>>3)*16+hk*8+(i&7)]
      smem[KF_OFF + ((wave * 2 + (nt >> 1)) * 32 + (l16 >> 3) * 16 + r) * FS + (l16 & 7) + (nt & 1) * 8] =
          (_Float16)acc[1][nt][j];
      // V -> B-frag layout: frag(jb,nt) element (lane=(hv,n), i) = v[jb*32+(i>>3)*16+hv*8+(i&7)][nt*16+n]
      smem[VF_OFF + (((wave >> 1) * 4 + nt) * 32 + (r >> 3) * 16 + l16) * FS + (wave & 1) * 8 + (r & 7)] =
          (_Float16)acc[2][nt][j];
    }
  __syncthreads();   // the only workgroup barrier in the kernel

  // ---------------- Phase 2: causal flash attention over LDS-resident frags ----------
  v16h qa0 = load_frag(smem + QF_OFF + ((wave * 2 + 0) * 32 + lane) * FS);
  v16h qa1 = load_frag(smem + QF_OFF + ((wave * 2 + 1) * 32 + lane) * FS);

  _Float16* ps = smem + PS_OFF + wave * 32 * FS;  // per-wave P scratch (A-frag layout)

  v8f O[4];
#pragma unroll
  for (int nt = 0; nt < 4; ++nt) O[nt] = v8f_zero();
  float mrow[8], lrow[8];
#pragma unroll
  for (int j = 0; j < 8; ++j) { mrow[j] = -INFINITY; lrow[j] = 0.0f; }

  const int jbmax = wave >> 1;  // causal: only key blocks up to the diagonal
  for (int jb = 0; jb <= jbmax; ++jb) {
    const int s0 = jb * 32;

    // S[16x32] = Q(16x64) @ K^T : two 16x16 accumulators, 2 K-steps each
    v8f S0 = v8f_zero(), S1 = v8f_zero();
    {
      v16h k00 = load_frag(smem + KF_OFF + (((jb * 2 + 0) * 2 + 0) * 32 + lane) * FS);
      v16h k01 = load_frag(smem + KF_OFF + (((jb * 2 + 0) * 2 + 1) * 32 + lane) * FS);
      v16h k10 = load_frag(smem + KF_OFF + (((jb * 2 + 1) * 2 + 0) * 32 + lane) * FS);
      v16h k11 = load_frag(smem + KF_OFF + (((jb * 2 + 1) * 2 + 1) * 32 + lane) * FS);
      S0 = __builtin_amdgcn_wmma_f32_16x16x32_f16(false, qa0, false, k00, (short)0, S0, false, false);
      S0 = __builtin_amdgcn_wmma_f32_16x16x32_f16(false, qa1, false, k01, (short)0, S0, false, false);
      S1 = __builtin_amdgcn_wmma_f32_16x16x32_f16(false, qa0, false, k10, (short)0, S1, false, false);
      S1 = __builtin_amdgcn_wmma_f32_16x16x32_f16(false, qa1, false, k11, (short)0, S1, false, false);
    }

    // online softmax; rows live in 16-lane half-groups
#pragma unroll
    for (int j = 0; j < 8; ++j) {
      int t  = wave * 16 + j + half * 8;
      float v0 = ((s0 + l16) <= t)      ? S0[j] * ATTN_SCALE : -INFINITY;
      float v1 = ((s0 + 16 + l16) <= t) ? S1[j] * ATTN_SCALE : -INFINITY;

      float rm = fmaxf(v0, v1);
#pragma unroll
      for (int off = 1; off < 16; off <<= 1)
        rm = fmaxf(rm, __shfl_xor(rm, off, 32));
      float mnew = fmaxf(mrow[j], rm);

      float p0 = __expf(v0 - mnew);
      float p1 = __expf(v1 - mnew);
      float rs = p0 + p1;
#pragma unroll
      for (int off = 1; off < 16; off <<= 1)
        rs += __shfl_xor(rs, off, 32);

      float alpha = __expf(mrow[j] - mnew);
      lrow[j] = lrow[j] * alpha + rs;
      mrow[j] = mnew;
#pragma unroll
      for (int nt = 0; nt < 4; ++nt) O[nt][j] *= alpha;

      // scatter P into A-frag layout; same-wave DS ops are in-order -> no barrier
      int lp = (l16 >> 3) * 16 + (j + 8 * half);   // target lane-slot (row = j+8*half)
      ps[lp * FS + (l16 & 7)]     = (_Float16)p0;  // key col l16    -> i = (l16&7)
      ps[lp * FS + (l16 & 7) + 8] = (_Float16)p1;  // key col 16+l16 -> i = (l16&7)+8
    }

    // O += P(16x32) @ V(32x64)
    v16h pa = load_frag(ps + lane * FS);
#pragma unroll
    for (int nt = 0; nt < 4; ++nt) {
      v16h bv = load_frag(smem + VF_OFF + ((jb * 4 + nt) * 32 + lane) * FS);
      O[nt] = __builtin_amdgcn_wmma_f32_16x16x32_f16(
          false, pa, false, bv, (short)0, O[nt], false, false);
    }
  }

  // ---------------- epilogue: out = O / l ----------------
  float* ob = out + (size_t)b * TT * HH;
#pragma unroll
  for (int nt = 0; nt < 4; ++nt)
#pragma unroll
    for (int j = 0; j < 8; ++j) {
      int t = wave * 16 + j + half * 8;
      ob[t * HH + nt * 16 + l16] = O[nt][j] / lrow[j];
    }
}

extern "C" void kernel_launch(void* const* d_in, const int* in_sizes, int n_in,
                              void* d_out, int out_size, void* d_ws, size_t ws_size,
                              hipStream_t stream) {
  (void)in_sizes; (void)n_in; (void)out_size; (void)ws_size;
  const float* x  = (const float*)d_in[0];
  const float* Wq = (const float*)d_in[1];
  const float* Wk = (const float*)d_in[2];
  const float* Wv = (const float*)d_in[3];
  float* outp = (float*)d_out;
  _Float16* wfrag = (_Float16*)d_ws;

  // 1) materialize the W fragment image (147 KB, L2-resident for the main kernel)
  build_wfrags<<<dim3((WF_TOTAL + 255) / 256), dim3(256), 0, stream>>>(Wq, Wk, Wv, wfrag);

  // 2) fused qkv-projection + causal flash attention
  hipFuncSetAttribute((const void*)attn_head_fused,
                      hipFuncAttributeMaxDynamicSharedMemorySize,
                      (int)SMEM_BYTES);
  attn_head_fused<<<dim3(BB), dim3(NTHREADS), SMEM_BYTES, stream>>>(x, wfrag, outp);
}
